// MultiViewAttentionFusion_32238024524165
// MI455X (gfx1250) — matrix operand order repaired and checked by reading, not compiled, not run on hardware
//
#include <hip/hip_runtime.h>

// ---------------- types ----------------
typedef __bf16 bf16;
typedef __attribute__((ext_vector_type(2)))  __bf16 bf16x2;
typedef __attribute__((ext_vector_type(8)))  __bf16 bf16x8;
typedef __attribute__((ext_vector_type(16))) __bf16 v16bf;
typedef __attribute__((ext_vector_type(8)))  float  v8f;

// ---------------- problem constants ----------------
#define N_   4
#define V_   4
#define C_   64
#define HW_  36864            // 192*192
#define TOK_ (N_ * HW_)       // 147456 tokens
#define PPW  8                // pixels per wave (32 rows = two 16-row M-tiles)
#define WPB  2                // waves per block
#define TPB  (WPB * 32)

#define XS  72                // bf16 row stride (64+8) -> conflict-free ds_load_b128
#define VTS 32                // s_uT row stride (o-major, K slots = 32 rows, all valid)
#define SS  18                // f32 sim row stride
#define AS  40                // bf16 attn A-tile row stride (32 K slots + pad)
#define LTS 36                // f32 logits-transposed row stride (32 rows + pad)

// ---- WMMA fragment loaders (CDNA5 16-bit layouts, wave32) ----
// A 16x32: lanes 0-15 row m hold K=0..7,16..23 ; lanes 16-31 hold K=8..15,24..31
__device__ __forceinline__ v16bf load_afrag(const bf16* rowbase, int kofs, int hi) {
    union { v16bf v; bf16x8 h[2]; } u;
    u.h[0] = *(const bf16x8*)(rowbase + kofs + hi * 8);
    u.h[1] = *(const bf16x8*)(rowbase + kofs + 16 + hi * 8);
    return u.v;
}
// B 32x16: lanes 0-15 col n hold K=0..7,16..23 ; lanes 16-31 hold K=8..15,24..31
__device__ __forceinline__ v16bf load_bfrag(const bf16* colbase, int kofs, int hi) {
    union { v16bf v; bf16x8 h[2]; } u;
    u.h[0] = *(const bf16x8*)(colbase + kofs + hi * 8);
    u.h[1] = *(const bf16x8*)(colbase + kofs + 16 + hi * 8);
    return u.v;
}

#define WMMA_BF16(A, B, Cacc) \
    __builtin_amdgcn_wmma_f32_16x16x32_bf16(false, (A), false, (B), (short)0, (Cacc), false, false)

// ------- prep: fuse weight pairs into two 64x64 bf16 matrices in d_ws -------
// mt[a][b] = sum_o Wq[o][b]*Wk[o][a]   (B-col a of (Wq^T Wk) = row a of mt)
// pt[a][b] = sum_t Wv[t][b]*Wo[a][t]   (B-col a of (Wv^T Wo^T) = row a of pt)
__global__ void mvaf_prep(const float* __restrict__ wq, const float* __restrict__ wk,
                          const float* __restrict__ wv, const float* __restrict__ wo,
                          bf16* __restrict__ ws) {
    int idx = blockIdx.x * 256 + threadIdx.x;     // 0 .. 8191
    int which = idx >> 12;
    int a = (idx >> 6) & 63, b = idx & 63;
    float acc = 0.0f;
    if (which == 0) {
        for (int o = 0; o < C_; ++o) acc += wq[o * C_ + b] * wk[o * C_ + a];
    } else {
        for (int t = 0; t < C_; ++t) acc += wv[t * C_ + b] * wo[a * C_ + t];
    }
    ws[idx] = (bf16)acc;
}

// ---------------- fused multi-view attention kernel ----------------
__global__ __launch_bounds__(TPB) void mvaf_kernel(const float* __restrict__ x,
                                                   const float* __restrict__ dep,
                                                   const bf16*  __restrict__ wfused,
                                                   float* __restrict__ out) {
    __shared__ __align__(16) bf16  s_x [WPB][32 * XS];     // xf tile: rows = pix*4+v
    __shared__ __align__(16) bf16  s_t [WPB][32 * XS];     // T = X*M rows (B for sim)
    __shared__ __align__(16) bf16  s_uT[WPB][64 * VTS];    // U = X*P transposed [o][row]
    __shared__ __align__(16) float s_sim[WPB][32 * SS];    // sim' rows (transposed sim)
    __shared__ __align__(16) bf16  s_at[WPB][32 * AS];     // block-diag attn A-tiles (32x32)
    __shared__ __align__(16) float s_lT[WPB][64 * LTS];    // logits transposed [o][row]
    __shared__ float s_m[WPB][32];                         // validity mask

    const int lane = threadIdx.x & 31;
    const int wv   = threadIdx.x >> 5;
    const int tile = blockIdx.x * WPB + wv;                // one wave = 8 pixels
    const int p0   = tile * PPW;
    const int n    = p0 / HW_;
    const int hw0  = p0 - n * HW_;                         // multiple of 8 -> float4 aligned

    // ---- stage 0: load x tile (4 views x 64 ch x 8 pixels) bf16-packed; depth mask ----
    const float* xb = x + (size_t)n * (V_ * C_ * HW_) + hw0;
    #pragma unroll
    for (int it = 0; it < 4; ++it) {
        int p = lane + (it << 5);                          // p = v*32 + c/2
        int v = p >> 5, c2 = (p & 31) << 1;
        const float* base = xb + (size_t)(v * C_ + c2) * HW_;
        float4 pa0 = *(const float4*)(base);               // ch c2,   pixels 0..3
        float4 pa1 = *(const float4*)(base + 4);           // ch c2,   pixels 4..7
        float4 pb0 = *(const float4*)(base + HW_);         // ch c2+1, pixels 0..3
        float4 pb1 = *(const float4*)(base + HW_ + 4);     // ch c2+1, pixels 4..7
        bf16x2 t;
        t.x=(bf16)pa0.x; t.y=(bf16)pb0.x; *(bf16x2*)&s_x[wv][(0*4+v)*XS + c2] = t;
        t.x=(bf16)pa0.y; t.y=(bf16)pb0.y; *(bf16x2*)&s_x[wv][(1*4+v)*XS + c2] = t;
        t.x=(bf16)pa0.z; t.y=(bf16)pb0.z; *(bf16x2*)&s_x[wv][(2*4+v)*XS + c2] = t;
        t.x=(bf16)pa0.w; t.y=(bf16)pb0.w; *(bf16x2*)&s_x[wv][(3*4+v)*XS + c2] = t;
        t.x=(bf16)pa1.x; t.y=(bf16)pb1.x; *(bf16x2*)&s_x[wv][(4*4+v)*XS + c2] = t;
        t.x=(bf16)pa1.y; t.y=(bf16)pb1.y; *(bf16x2*)&s_x[wv][(5*4+v)*XS + c2] = t;
        t.x=(bf16)pa1.z; t.y=(bf16)pb1.z; *(bf16x2*)&s_x[wv][(6*4+v)*XS + c2] = t;
        t.x=(bf16)pa1.w; t.y=(bf16)pb1.w; *(bf16x2*)&s_x[wv][(7*4+v)*XS + c2] = t;
    }
    {
        int pix = lane & 7, v = lane >> 3;
        float dv = dep[(size_t)(n * V_ + v) * HW_ + hw0 + pix];
        s_m[wv][pix * 4 + v] = (dv > 0.0f) ? 1.0f : 0.0f;
    }
    {   // zero attn A-tile (block-diagonal; finite values required by WMMA)
        bf16x8 z = {};
        *(bf16x8*)&s_at[wv][lane * AS + 0]  = z;
        *(bf16x8*)&s_at[wv][lane * AS + 8]  = z;
        *(bf16x8*)&s_at[wv][lane * AS + 16] = z;
        *(bf16x8*)&s_at[wv][lane * AS + 24] = z;
    }
    __syncthreads();

    // ---- stage 1: T = X*M, U = X*P  (two 16-row M-tiles share every B-fragment) ----
    const int hi = lane >> 4;
    const int nl = lane & 15;
    v16bf a0[2], a1[2];
    #pragma unroll
    for (int g = 0; g < 2; ++g) {
        const bf16* ar = &s_x[wv][(g * 16 + nl) * XS];
        a0[g] = load_afrag(ar, 0, hi);
        a1[g] = load_afrag(ar, 32, hi);
    }
    const bf16* mt = wfused;                               // 64x64 row-major (B columns)
    const bf16* pt = wfused + C_ * C_;
    #pragma unroll
    for (int nt = 0; nt < 4; ++nt) {
        int col = nt * 16 + nl;
        v16bf bm0 = load_bfrag(mt + col * C_, 0, hi);
        v16bf bm1 = load_bfrag(mt + col * C_, 32, hi);
        v16bf bp0 = load_bfrag(pt + col * C_, 0, hi);
        v16bf bp1 = load_bfrag(pt + col * C_, 32, hi);
        #pragma unroll
        for (int g = 0; g < 2; ++g) {
            v8f acct = {}, accu = {};
            acct = WMMA_BF16(a0[g], bm0, acct);
            acct = WMMA_BF16(a1[g], bm1, acct);
            accu = WMMA_BF16(a0[g], bp0, accu);
            accu = WMMA_BF16(a1[g], bp1, accu);
            #pragma unroll
            for (int j = 0; j < 8; ++j)                    // T row-major (B for sim)
                s_t[wv][(g * 16 + hi * 8 + j) * XS + col] = (bf16)acct[j];
            bf16x8 u8;                                     // U transposed: 1 packed store
            #pragma unroll
            for (int j = 0; j < 8; ++j) u8[j] = (bf16)accu[j];
            *(bf16x8*)&s_uT[wv][col * VTS + g * 16 + hi * 8] = u8;
        }
    }
    __syncthreads();

    // ---- stage 2a: sim'_g = X_g @ T_g^T  (per M-tile; sim'[m][n] = sim[n][m]) ----
    #pragma unroll
    for (int g = 0; g < 2; ++g) {
        v16bf bt0 = load_bfrag(&s_t[wv][(g * 16 + nl) * XS], 0, hi);
        v16bf bt1 = load_bfrag(&s_t[wv][(g * 16 + nl) * XS], 32, hi);
        v8f sim = {};
        sim = WMMA_BF16(a0[g], bt0, sim);
        sim = WMMA_BF16(a1[g], bt1, sim);
        #pragma unroll
        for (int j = 0; j < 8; ++j)
            s_sim[wv][(g * 16 + hi * 8 + j) * SS + nl] = sim[j];
    }
    __syncthreads();

    // ---- stage 2b: 4-way softmax per row (all 32 lanes) -> attn A-tile ----
    {
        int m = lane, tok = m >> 2, lc = m & 15;           // sim[m][tok*4+j] = sim'[tok*4+j][m%16]
        float l0 = s_sim[wv][(tok * 4 + 0) * SS + lc];
        float l1 = s_sim[wv][(tok * 4 + 1) * SS + lc];
        float l2 = s_sim[wv][(tok * 4 + 2) * SS + lc];
        float l3 = s_sim[wv][(tok * 4 + 3) * SS + lc];
        float e1 = __expf(l1 - l0), e2 = __expf(l2 - l0), e3 = __expf(l3 - l0);
        float inv = __frcp_rn(1.0f + e1 + e2 + e3);
        bf16* ar = &s_at[wv][m * AS + tok * 4];            // K index = global row tok*4+j
        ar[0] = (bf16)inv;        ar[1] = (bf16)(e1 * inv);
        ar[2] = (bf16)(e2 * inv); ar[3] = (bf16)(e3 * inv);
    }
    __syncthreads();

    // ---- stage 2c: logits = Attn(32x32 blockdiag) @ U -> transposed f32 tile ----
    #pragma unroll
    for (int g = 0; g < 2; ++g) {
        v16bf aat = load_afrag(&s_at[wv][(g * 16 + nl) * AS], 0, hi);
        #pragma unroll
        for (int nt = 0; nt < 4; ++nt) {
            int col = nt * 16 + nl;
            v16bf bu = load_bfrag(&s_uT[wv][col * VTS], 0, hi);
            v8f acc = {};
            acc = WMMA_BF16(aat, bu, acc);
            *(float4*)&s_lT[wv][col * LTS + g * 16 + hi * 8]     = make_float4(acc[0], acc[1], acc[2], acc[3]);
            *(float4*)&s_lT[wv][col * LTS + g * 16 + hi * 8 + 4] = make_float4(acc[4], acc[5], acc[6], acc[7]);
        }
    }
    __syncthreads();

    // ---- stage 3: view softmax + mask + renormalize + fuse + reduce + store ----
    // res = sum_v(e_v*m_v*x_v) / (sum_v(e_v*m_v) + 1e-7*sum_v(e_v))   [one division]
    float* ob = out + (size_t)n * (C_ * HW_) + hw0;
    const int c = lane << 1;                               // channels c, c+1
    float r0[8], r1[8];
    #pragma unroll
    for (int pix = 0; pix < 8; ++pix) {
        float m0 = s_m[wv][pix * 4 + 0], m1 = s_m[wv][pix * 4 + 1];
        float m2 = s_m[wv][pix * 4 + 2], m3 = s_m[wv][pix * 4 + 3];
        bf16x2 x0 = *(const bf16x2*)&s_x[wv][(pix * 4 + 0) * XS + c];
        bf16x2 x1 = *(const bf16x2*)&s_x[wv][(pix * 4 + 1) * XS + c];
        bf16x2 x2 = *(const bf16x2*)&s_x[wv][(pix * 4 + 2) * XS + c];
        bf16x2 x3 = *(const bf16x2*)&s_x[wv][(pix * 4 + 3) * XS + c];
        {
            float4 lv = *(const float4*)&s_lT[wv][c * LTS + pix * 4];
            float e1 = __expf(lv.y - lv.x), e2 = __expf(lv.z - lv.x), e3 = __expf(lv.w - lv.x);
            float S   = 1.0f + e1 + e2 + e3;
            float em0 = m0, em1 = e1 * m1, em2 = e2 * m2, em3 = e3 * m3;
            float num = em0 * (float)x0.x + em1 * (float)x1.x + em2 * (float)x2.x + em3 * (float)x3.x;
            r0[pix] = num * __frcp_rn(em0 + em1 + em2 + em3 + 1e-7f * S);
        }
        {
            float4 lv = *(const float4*)&s_lT[wv][(c + 1) * LTS + pix * 4];
            float e1 = __expf(lv.y - lv.x), e2 = __expf(lv.z - lv.x), e3 = __expf(lv.w - lv.x);
            float S   = 1.0f + e1 + e2 + e3;
            float em0 = m0, em1 = e1 * m1, em2 = e2 * m2, em3 = e3 * m3;
            float num = em0 * (float)x0.y + em1 * (float)x1.y + em2 * (float)x2.y + em3 * (float)x3.y;
            r1[pix] = num * __frcp_rn(em0 + em1 + em2 + em3 + 1e-7f * S);
        }
    }
    *(float4*)(ob + (size_t)c * HW_)           = make_float4(r0[0], r0[1], r0[2], r0[3]);
    *(float4*)(ob + (size_t)c * HW_ + 4)       = make_float4(r0[4], r0[5], r0[6], r0[7]);
    *(float4*)(ob + (size_t)(c + 1) * HW_)     = make_float4(r1[0], r1[1], r1[2], r1[3]);
    *(float4*)(ob + (size_t)(c + 1) * HW_ + 4) = make_float4(r1[4], r1[5], r1[6], r1[7]);
}

// ---------------- launcher ----------------
extern "C" void kernel_launch(void* const* d_in, const int* in_sizes, int n_in,
                              void* d_out, int out_size, void* d_ws, size_t ws_size,
                              hipStream_t stream) {
    (void)in_sizes; (void)n_in; (void)out_size; (void)ws_size;
    const float* x  = (const float*)d_in[0];
    const float* dd = (const float*)d_in[1];
    const float* wq = (const float*)d_in[2];
    const float* wk = (const float*)d_in[3];
    const float* wv = (const float*)d_in[4];
    const float* wo = (const float*)d_in[5];
    bf16* wfused = (bf16*)d_ws;                            // 2 * 64*64 bf16 = 16 KB

    mvaf_prep<<<32, 256, 0, stream>>>(wq, wk, wv, wo, wfused);

    const int blocks = TOK_ / (PPW * WPB);                 // 147456 / 16 = 9216
    mvaf_kernel<<<blocks, TPB, 0, stream>>>(x, dd, wfused, (float*)d_out);
}
